// LSTMPredictor_36481452212835
// MI455X (gfx1250) — matrix operand order repaired and checked by reading, not compiled, not run on hardware
//
#include <hip/hip_runtime.h>

// CDNA5 / gfx1250, wave32. One wave owns 16 batch rows of the LSTM.
// Per step: gates^T[64x16] = Whh(A,f16) x h^T(B,f16) + xproj(C,f32)
// computed as 4x v_wmma_f32_16x16x32_f16 (K=16 zero-padded to 32).

typedef __attribute__((ext_vector_type(16))) _Float16 v16h;
typedef __attribute__((ext_vector_type(8)))  float    v8f;

#define HIDDEN 16
#define TSTEPS 2048

__device__ __forceinline__ float fast_sigmoid(float z) {
    // 1/(1+exp(-z)) via v_exp_f32 (exp2) + v_rcp_f32
    return __builtin_amdgcn_rcpf(1.0f + __builtin_amdgcn_exp2f(-1.44269504f * z));
}
__device__ __forceinline__ float fast_tanh(float z) {
    // tanh(z) = 1 - 2/(exp2(2z*log2e)+1)
    return 1.0f - 2.0f * __builtin_amdgcn_rcpf(1.0f + __builtin_amdgcn_exp2f(2.88539008f * z));
}

__global__ __launch_bounds__(64) void lstm_wmma_kernel(
    const float* __restrict__ x,     // [B, T, 1]
    const float* __restrict__ w_ih,  // [64, 1]
    const float* __restrict__ w_hh,  // [64, 16]
    const float* __restrict__ b_ih,  // [64]
    const float* __restrict__ b_hh,  // [64]
    const float* __restrict__ fc_w,  // [1, 16]
    const float* __restrict__ fc_b,  // [1]
    float* __restrict__ out)         // [B, 1]
{
    const int lane = threadIdx.x & 31;
    const int half = lane >> 4;      // 0: lanes 0-15, 1: lanes 16-31
    const int mrow = lane & 15;
    const int wave = blockIdx.x * (blockDim.x >> 5) + (threadIdx.x >> 5);
    const int batchBase = wave * 16; // this wave's 16 batch rows (N dimension)

    // ---- A matrices: Whh per gate block, f16, 16x32 layout (K 16..31 = 0) ----
    // A element i (i<8) of lane = Whh[16*gb + mrow][8*half + i]
    v16h a[4];
    float wih_c[4][8];
    float bias_c[4][8];
#pragma unroll
    for (int gb = 0; gb < 4; ++gb) {
        const float* row = w_hh + (gb * 16 + mrow) * HIDDEN + half * 8;
#pragma unroll
        for (int i = 0; i < 8; ++i) a[gb][i] = (_Float16)row[i];
#pragma unroll
        for (int i = 8; i < 16; ++i) a[gb][i] = (_Float16)0.0f;
        // D layout: lane, vgpr r -> gate g = 16*gb + r + 8*half
        const int gBase = gb * 16 + half * 8;
#pragma unroll
        for (int r = 0; r < 8; ++r) {
            wih_c[gb][r]  = w_ih[gBase + r];
            bias_c[gb][r] = b_ih[gBase + r] + b_hh[gBase + r];
        }
    }

    // ---- B matrix (h^T, f16 32x16): lanes 0-15 hold K=0..7, lanes 16-31 K=8..15; elements 8..15 are K=16..31 (=0 pad)
    v16h bh;
#pragma unroll
    for (int i = 0; i < 16; ++i) bh[i] = (_Float16)0.0f;

    // ---- c, h state in f32 C/D layout: lane -> batch = lane&15, vgpr r -> hidden r+8*half
    float cst[8];
    float hv[8];
#pragma unroll
    for (int r = 0; r < 8; ++r) { cst[r] = 0.0f; hv[r] = 0.0f; }

    const float4* xp = (const float4*)(x + (size_t)(batchBase + mrow) * TSTEPS);
    const bool isLow = (half == 0);

    for (int t4 = 0; t4 < TSTEPS / 4; ++t4) {
        const float4 xq = xp[t4];     // 4 timesteps of this lane's batch column
        float xs[4] = {xq.x, xq.y, xq.z, xq.w};
#pragma unroll
        for (int u = 0; u < 4; ++u) {
            const float xv = xs[u];
            v8f acc0, acc1, acc2, acc3;
#pragma unroll
            for (int r = 0; r < 8; ++r) {
                acc0[r] = fmaf(xv, wih_c[0][r], bias_c[0][r]);
                acc1[r] = fmaf(xv, wih_c[1][r], bias_c[1][r]);
                acc2[r] = fmaf(xv, wih_c[2][r], bias_c[2][r]);
                acc3[r] = fmaf(xv, wih_c[3][r], bias_c[3][r]);
            }
            acc0 = __builtin_amdgcn_wmma_f32_16x16x32_f16(false, a[0], false, bh, (short)0, acc0, false, false);
            acc1 = __builtin_amdgcn_wmma_f32_16x16x32_f16(false, a[1], false, bh, (short)0, acc1, false, false);
            acc2 = __builtin_amdgcn_wmma_f32_16x16x32_f16(false, a[2], false, bh, (short)0, acc2, false, false);
            acc3 = __builtin_amdgcn_wmma_f32_16x16x32_f16(false, a[3], false, bh, (short)0, acc3, false, false);

#pragma unroll
            for (int r = 0; r < 8; ++r) {
                const float iv = fast_sigmoid(acc0[r]);
                const float fv = fast_sigmoid(acc1[r]);
                const float gv = fast_tanh(acc2[r]);
                const float ov = fast_sigmoid(acc3[r]);
                const float cn = fmaf(fv, cst[r], iv * gv);
                cst[r] = cn;                 // c stays fp32 across all steps
                hv[r]  = ov * fast_tanh(cn);
            }
            // Rebuild B from h: lane half h holds hidden 8*half+r in element r
            // (table order, matching A); elements 8..15 are the zero K pad.
#pragma unroll
            for (int r = 0; r < 8; ++r) {
                const float other = __shfl_xor(hv[r], 16, 32);
                bh[r]     = (_Float16)hv[r];
                bh[8 + r] = (_Float16)0.0f;
            }
        }
    }

    // ---- head: out[b] = sigmoid(sum_k h[b,k]*fc_w[k] + fc_b) ----
    float partial = 0.0f;
#pragma unroll
    for (int r = 0; r < 8; ++r)
        partial = fmaf(hv[r], fc_w[half * 8 + r], partial);
    partial += __shfl_xor(partial, 16, 32);   // combine hidden 0-7 and 8-15 halves
    if (lane < 16)
        out[batchBase + lane] = fast_sigmoid(partial + fc_b[0]);
}

extern "C" void kernel_launch(void* const* d_in, const int* in_sizes, int n_in,
                              void* d_out, int out_size, void* d_ws, size_t ws_size,
                              hipStream_t stream) {
    const float* x    = (const float*)d_in[0];
    const float* w_ih = (const float*)d_in[1];
    const float* w_hh = (const float*)d_in[2];
    const float* b_ih = (const float*)d_in[3];
    const float* b_hh = (const float*)d_in[4];
    const float* fc_w = (const float*)d_in[5];
    const float* fc_b = (const float*)d_in[6];
    float* out = (float*)d_out;

    const int B = in_sizes[0] / TSTEPS;   // 8192
    const int waves = B / 16;             // 512 independent recurrences
    const int wavesPerBlock = 2;          // 64 threads/block -> spread over WGPs
    dim3 block(32 * wavesPerBlock);
    dim3 grid(waves / wavesPerBlock);
    lstm_wmma_kernel<<<grid, block, 0, stream>>>(x, w_ih, w_hh, b_ih, b_hh, fc_w, fc_b, out);
}
